// NewDnnPosTagger_77996606095699
// MI455X (gfx1250) — matrix operand order repaired and checked by reading, not compiled, not run on hardware
//
#include <hip/hip_runtime.h>
#include <hip/hip_bf16.h>

// ---------------- model dims ----------------
#define Hd    256
#define H2d   512
#define Ld    100
#define Bd    16
#define Md    (Bd * Ld)   // 1600 rows (b*100 + t)
#define G4H   1024        // 4*H gates
#define IN0P  160         // 150 padded to 160 (K % 32 == 0)
#define HC2   128         // H/2

typedef __attribute__((ext_vector_type(16))) _Float16 v16h;
typedef __attribute__((ext_vector_type(8)))  float     v8f;
typedef __attribute__((ext_vector_type(8)))  _Float16  h8;

// ---- WMMA fragment helpers (16x16x32 f16 -> f32) ----
// A (16xK tile, row-major): lane L holds row m=L%16; lane-group lg=L/16 selects K halves.
static __device__ inline v16h frag_a_g(const _Float16* __restrict__ Ab, int K, int kb, int lane) {
  int m = lane & 15, lg = lane >> 4;
  const _Float16* p = Ab + (size_t)m * K + kb + lg * 8;
  h8 lo = *(const h8*)p;
  h8 hi = *(const h8*)(p + 16);
  v16h a;
#pragma unroll
  for (int e = 0; e < 8; ++e) { a[e] = lo[e]; a[e + 8] = hi[e]; }
  return a;
}

// B = W^T where W is [N][K] row-major: lane L holds column n=L%16; kg=L/16 selects K half.
static __device__ inline v16h frag_b_g(const _Float16* __restrict__ Wb, int K, int kb, int lane) {
  int n = lane & 15, kg = lane >> 4;
  const _Float16* p = Wb + (size_t)n * K + kb + kg * 8;
  h8 lo = *(const h8*)p;
  h8 hi = *(const h8*)(p + 16);
  v16h b;
#pragma unroll
  for (int e = 0; e < 8; ++e) { b[e] = lo[e]; b[e + 8] = hi[e]; }
  return b;
}

static __device__ inline v8f wmma_f16(v16h a, v16h b, v8f c) {
  return __builtin_amdgcn_wmma_f32_16x16x32_f16(false, a, false, b, (short)0, c, false, false);
}

// ---------------- utility kernels ----------------
__global__ void zero_f32_kernel(float* __restrict__ p, long n) {
  long g = blockIdx.x * (long)blockDim.x + threadIdx.x;
  if (g < n) p[g] = 0.f;
}
__global__ void zero_f16_kernel(_Float16* __restrict__ p, long n) {
  long g = blockIdx.x * (long)blockDim.x + threadIdx.x;
  if (g < n) p[g] = (_Float16)0.f;
}

// f32 -> f16 submatrix convert with optional zero-padding of columns
__global__ void cvt_f16_kernel(const float* __restrict__ src, _Float16* __restrict__ dst,
                               int rows, int cols, int src_stride, int src_off, int dst_cols) {
  long g = blockIdx.x * (long)blockDim.x + threadIdx.x;
  long total = (long)rows * dst_cols;
  if (g >= total) return;
  int r = (int)(g / dst_cols), c = (int)(g % dst_cols);
  dst[g] = (c < cols) ? (_Float16)src[(size_t)r * src_stride + src_off + c] : (_Float16)0.f;
}

// x0[m][0:100]=word_embeds, [100:150]=pos_emb[pos_idx], [150:160]=0   (f16)
__global__ void embed_concat_kernel(const float* __restrict__ we, const int* __restrict__ pidx,
                                    const float* __restrict__ pemb, _Float16* __restrict__ x0) {
  long g = blockIdx.x * (long)blockDim.x + threadIdx.x;
  if (g >= (long)Md * IN0P) return;
  int m = (int)(g / IN0P), c = (int)(g % IN0P);
  float v = 0.f;
  if (c < 100)      v = we[(size_t)m * 100 + c];
  else if (c < 150) v = pemb[(size_t)pidx[m] * 50 + (c - 100)];
  x0[g] = (_Float16)v;
}

// ---------------- generic WMMA GEMM: C[M,N] = A[M,K] @ W[N,K]^T + bias0 + bias1 ----------------
// outmode 0: f32 out at row m.  1: f16 out at row m.  2: f16 out at row (m/100)*101 + m%100 + 1 (h_root remap)
__global__ __launch_bounds__(256) void gemm_wmma_kernel(
    const _Float16* __restrict__ A, const _Float16* __restrict__ W,
    const float* __restrict__ bias0, const float* __restrict__ bias1,
    float* __restrict__ Cf, _Float16* __restrict__ Ch,
    int M, int N, int K, int outmode) {
  int lane = threadIdx.x & 31;
  int w    = threadIdx.x >> 5;
  int mtile = blockIdx.x;
  int ntile = blockIdx.y * 8 + w;
  if (ntile * 16 >= N || mtile * 16 >= M) return;  // wave-uniform

  const _Float16* Ab = A + (size_t)mtile * 16 * K;
  const _Float16* Wb = W + (size_t)ntile * 16 * K;

  v8f acc;
#pragma unroll
  for (int v = 0; v < 8; ++v) acc[v] = 0.f;

  for (int kb = 0; kb < K; kb += 32) {
    if (kb + 32 < K)
      __builtin_prefetch((const void*)(Wb + (size_t)(lane & 15) * K + kb + 32), 0, 1);
    v16h a = frag_a_g(Ab, K, kb, lane);
    v16h b = frag_b_g(Wb, K, kb, lane);
    acc = wmma_f16(a, b, acc);
  }

  int n = ntile * 16 + (lane & 15);
  int half = lane >> 4;
  float bsum = (bias0 ? bias0[n] : 0.f) + (bias1 ? bias1[n] : 0.f);
#pragma unroll
  for (int v = 0; v < 8; ++v) {
    int m = mtile * 16 + v + 8 * half;
    float val = acc[v] + bsum;
    if (outmode == 0)      Cf[(size_t)m * N + n] = val;
    else if (outmode == 1) Ch[(size_t)m * N + n] = (_Float16)val;
    else {
      int r = (m / 100) * 101 + (m % 100) + 1;
      Ch[(size_t)r * N + n] = (_Float16)val;
    }
  }
}

// ---------------- persistent LSTM recurrent scan ----------------
// grid.x = 2 (dir), 512 threads. Per step: g = gx[t] + h @ whh^T via WMMA (M=16 batch),
// then elementwise cell update; h, c live in LDS across all 100 steps.
__global__ __launch_bounds__(512) void lstm_scan_kernel(
    const float* __restrict__ gx_f, const float* __restrict__ gx_b,
    const _Float16* __restrict__ whh_f, const _Float16* __restrict__ whh_b,
    _Float16* __restrict__ xnext /* [1600][512] f16 */) {
  extern __shared__ char smem[];
  _Float16* sh_h = (_Float16*)smem;                 // 16*256 f16  = 8 KB
  float*    sh_c = (float*)(smem + 8192);           // 16*256 f32  = 16 KB
  float*    sh_g = (float*)(smem + 8192 + 16384);   // 16*1024 f32 = 64 KB

  const int dir = blockIdx.x;
  const float*    gx  = dir ? gx_b  : gx_f;
  const _Float16* whh = dir ? whh_b : whh_f;
  const int diroff = dir * Hd;

  int tid  = threadIdx.x;
  int lane = tid & 31;
  int w    = tid >> 5;              // 16 waves

  for (int idx = tid; idx < Bd * Hd; idx += 512) { sh_h[idx] = (_Float16)0.f; sh_c[idx] = 0.f; }
  __syncthreads();

  for (int s = 0; s < Ld; ++s) {
    int t = dir ? (Ld - 1 - s) : s;

    // ---- g = h @ whh^T : 64 N-tiles, 4 per wave ----
#pragma unroll
    for (int tp = 0; tp < 4; ++tp) {
      int ntile = w * 4 + tp;
      const _Float16* Wb = whh + (size_t)ntile * 16 * Hd;
      v8f acc;
#pragma unroll
      for (int v = 0; v < 8; ++v) acc[v] = 0.f;
#pragma unroll
      for (int kt = 0; kt < Hd / 32; ++kt) {
        int kb = kt * 32;
        if (kb + 32 < Hd)
          __builtin_prefetch((const void*)(Wb + (size_t)(lane & 15) * Hd + kb + 32), 0, 1);
        // A fragment from LDS h (16x256 f16)
        int m = lane & 15, lg = lane >> 4;
        const _Float16* p = sh_h + m * Hd + kb + lg * 8;
        h8 lo = *(const h8*)p;
        h8 hi = *(const h8*)(p + 16);
        v16h a;
#pragma unroll
        for (int e = 0; e < 8; ++e) { a[e] = lo[e]; a[e + 8] = hi[e]; }
        v16h bf = frag_b_g(Wb, Hd, kb, lane);
        acc = wmma_f16(a, bf, acc);
      }
      int n = ntile * 16 + (lane & 15);
      int half = lane >> 4;
#pragma unroll
      for (int v = 0; v < 8; ++v) {
        int bb = v + 8 * half;  // batch row
        sh_g[bb * G4H + n] = acc[v] + gx[((size_t)(bb * Ld + t)) * G4H + n];
      }
    }
    __syncthreads();

    // ---- cell update: 4096 (b,j) elements, 8 per thread ----
#pragma unroll
    for (int e = 0; e < 8; ++e) {
      int idx = e * 512 + tid;
      int bb = idx >> 8, j = idx & 255;
      const float* gr = sh_g + bb * G4H;
      float ig = gr[j], fg = gr[Hd + j], gg = gr[2 * Hd + j], og = gr[3 * Hd + j];
      float si = 1.f / (1.f + __expf(-ig));
      float sf = 1.f / (1.f + __expf(-fg));
      float so = 1.f / (1.f + __expf(-og));
      float c = sf * sh_c[idx] + si * tanhf(gg);
      float h = so * tanhf(c);
      sh_c[idx] = c;
      sh_h[idx] = (_Float16)h;
      xnext[((size_t)(bb * Ld + t)) * H2d + diroff + j] = (_Float16)h;
    }
    __syncthreads();
  }
}

// ---------------- fused biaffine scorer ----------------
// One block per (i, b). 7 waves; wave w owns j-tile [16w, 16w+16).
// h1 = relu(h_first[b,i] + h_root[b,j] + b1) built in-register as the WMMA A fragment,
// h2 = h1 @ W2^T (512->128) in WMMA, then relu+b2, dot W3, +b3, mask, store.
__global__ __launch_bounds__(224) void biaffine_score_kernel(
    const _Float16* __restrict__ h_first, const _Float16* __restrict__ h_root,
    const float* __restrict__ b1, const float* __restrict__ b2,
    const _Float16* __restrict__ w3, const float* __restrict__ b3,
    const int* __restrict__ sen_lens, const _Float16* __restrict__ W2f,
    float* __restrict__ out) {
  __shared__ float hf1[H2d];           // h_first row + b1 (f32)
  __shared__ float sred[7][16][16];    // per-wave reduction

  int i = blockIdx.x, b = blockIdx.y;
  int tid = threadIdx.x;

  for (int k = tid; k < H2d; k += 224)
    hf1[k] = (float)h_first[((size_t)(b * Ld + i)) * H2d + k] + b1[k];
  __syncthreads();

  int w = tid >> 5, lane = tid & 31;
  int m = lane & 15, lg = lane >> 4;
  int jbase = w * 16;
  int j  = jbase + m;
  int jj = (j <= Ld) ? j : 0;  // clamp pad rows into a valid row (results discarded)
  const _Float16* hr = h_root + ((size_t)(b * 101 + jj)) * H2d;

  v8f acc[8];
#pragma unroll
  for (int q = 0; q < 8; ++q)
#pragma unroll
    for (int v = 0; v < 8; ++v) acc[q][v] = 0.f;

  for (int kt = 0; kt < H2d / 32; ++kt) {
    int kb = kt * 32;
    int k0 = kb + lg * 8;
    h8 r0 = *(const h8*)(hr + k0);
    h8 r1 = *(const h8*)(hr + k0 + 16);
    v16h a;
#pragma unroll
    for (int e = 0; e < 8; ++e) {
      float v0 = (float)r0[e] + hf1[k0 + e];
      float v1 = (float)r1[e] + hf1[k0 + 16 + e];
      a[e]     = (_Float16)(v0 > 0.f ? v0 : 0.f);
      a[e + 8] = (_Float16)(v1 > 0.f ? v1 : 0.f);
    }
#pragma unroll
    for (int q = 0; q < 8; ++q) {
      v16h bf = frag_b_g(W2f + (size_t)q * 16 * H2d, H2d, kb, lane);
      acc[q] = wmma_f16(a, bf, acc[q]);
    }
  }

  // relu(h2 + b2) dot w3, reduced across the 16 n-columns via LDS
  int ncol = lane & 15, half = lane >> 4;
  float ps[8];
#pragma unroll
  for (int v = 0; v < 8; ++v) ps[v] = 0.f;
#pragma unroll
  for (int q = 0; q < 8; ++q) {
    int n = q * 16 + ncol;
    float w3v = (float)w3[n], b2v = b2[n];
#pragma unroll
    for (int v = 0; v < 8; ++v) {
      float r = acc[q][v] + b2v;
      ps[v] += w3v * (r > 0.f ? r : 0.f);
    }
  }
#pragma unroll
  for (int v = 0; v < 8; ++v) sred[w][8 * half + v][ncol] = ps[v];
  __syncthreads();

  if (lane < 16) {
    float s = b3[0];
#pragma unroll
    for (int c = 0; c < 16; ++c) s += sred[w][lane][c];
    int jo = jbase + lane;
    if (jo <= Ld) {
      int sen = sen_lens[b];
      bool ok = (i < sen) && (jo <= sen);
      float val = ok ? s : 0.f;
      if (i == 0 && jo == 0) val = 0.f;
      out[((size_t)b * 101 + i) * 101 + jo] = val;
    }
  }
}

// ---------------- host launcher ----------------
extern "C" void kernel_launch(void* const* d_in, const int* in_sizes, int n_in,
                              void* d_out, int out_size, void* d_ws, size_t ws_size,
                              hipStream_t stream) {
  (void)in_sizes; (void)n_in; (void)out_size; (void)ws_size;

  const float* word_embeds = (const float*)d_in[0];   // (16,100,100)
  const int*   pos_idx     = (const int*)  d_in[1];   // (16,100)
  const int*   sen_lens    = (const int*)  d_in[2];   // (16,)
  const float* pos_emb     = (const float*)d_in[3];   // (46,50)
  const float* w_ih0       = (const float*)d_in[4];   // (2,1024,150)
  const float* w_hh0       = (const float*)d_in[5];   // (2,1024,256)
  const float* b_ih0       = (const float*)d_in[6];   // (2,1024)
  const float* b_hh0       = (const float*)d_in[7];   // (2,1024)
  const float* w_ih        = (const float*)d_in[8];   // (3,2,1024,512)
  const float* w_hh        = (const float*)d_in[9];   // (3,2,1024,256)
  const float* b_ih        = (const float*)d_in[10];  // (3,2,1024)
  const float* b_hh        = (const float*)d_in[11];  // (3,2,1024)
  const float* W1          = (const float*)d_in[12];  // (512,1024)
  const float* b1          = (const float*)d_in[13];  // (512,)
  const float* W2          = (const float*)d_in[14];  // (128,512)
  const float* b2          = (const float*)d_in[15];  // (128,)
  const float* W3          = (const float*)d_in[16];  // (1,128)
  const float* b3          = (const float*)d_in[17];  // (1,)
  float* out = (float*)d_out;                          // (16,101,101)

  // -------- workspace carve --------
  char* wp = (char*)d_ws;
  auto alloc = [&](size_t bytes) -> char* {
    char* r = wp; wp += (bytes + 255) & ~(size_t)255; return r;
  };
  _Float16* x0     = (_Float16*)alloc((size_t)Md * IN0P * 2);
  _Float16* xb0    = (_Float16*)alloc((size_t)Md * H2d * 2);
  _Float16* xb1    = (_Float16*)alloc((size_t)Md * H2d * 2);
  _Float16* wih0h  = (_Float16*)alloc((size_t)2 * G4H * IN0P * 2);
  _Float16* whh0h  = (_Float16*)alloc((size_t)2 * G4H * Hd * 2);
  _Float16* wihh   = (_Float16*)alloc((size_t)6 * G4H * H2d * 2);
  _Float16* whhh   = (_Float16*)alloc((size_t)6 * G4H * Hd * 2);
  _Float16* W1a    = (_Float16*)alloc((size_t)H2d * H2d * 2);
  _Float16* W1b    = (_Float16*)alloc((size_t)H2d * H2d * 2);
  _Float16* W2h    = (_Float16*)alloc((size_t)HC2 * H2d * 2);
  _Float16* W3h    = (_Float16*)alloc(256);
  float*    gxF    = (float*)alloc((size_t)Md * G4H * 4);
  float*    gxB    = (float*)alloc((size_t)Md * G4H * 4);
  _Float16* hfirst = (_Float16*)alloc((size_t)Md * H2d * 2);
  _Float16* hroot  = (_Float16*)alloc((size_t)Bd * 101 * H2d * 2);

  auto blocks = [](long n) { return dim3((unsigned)((n + 255) / 256)); };

  // -------- init / precision conversion --------
  zero_f32_kernel<<<blocks(16L * 101 * 101), 256, 0, stream>>>(out, 16L * 101 * 101);
  cvt_f16_kernel<<<blocks(2048L * IN0P), 256, 0, stream>>>(w_ih0, wih0h, 2048, 150, 150, 0, IN0P);
  cvt_f16_kernel<<<blocks(2048L * Hd),   256, 0, stream>>>(w_hh0, whh0h, 2048, Hd, Hd, 0, Hd);
  cvt_f16_kernel<<<blocks(6144L * H2d),  256, 0, stream>>>(w_ih,  wihh,  6144, H2d, H2d, 0, H2d);
  cvt_f16_kernel<<<blocks(6144L * Hd),   256, 0, stream>>>(w_hh,  whhh,  6144, Hd, Hd, 0, Hd);
  cvt_f16_kernel<<<blocks(512L * H2d),   256, 0, stream>>>(W1, W1a, H2d, H2d, 2 * H2d, 0,   H2d);
  cvt_f16_kernel<<<blocks(512L * H2d),   256, 0, stream>>>(W1, W1b, H2d, H2d, 2 * H2d, H2d, H2d);
  cvt_f16_kernel<<<blocks(128L * H2d),   256, 0, stream>>>(W2, W2h, HC2, H2d, H2d, 0, H2d);
  cvt_f16_kernel<<<blocks(128L),         256, 0, stream>>>(W3, W3h, 1, HC2, HC2, 0, HC2);
  embed_concat_kernel<<<blocks((long)Md * IN0P), 256, 0, stream>>>(word_embeds, pos_idx, pos_emb, x0);

  const size_t scanLds = 8192 + 16384 + 65536;  // 90112 B (within 320 KB WGP LDS)

  // -------- layer 0 (K = 160 padded) --------
  gemm_wmma_kernel<<<dim3(Md / 16, G4H / 128), 256, 0, stream>>>(
      x0, wih0h, b_ih0, b_hh0, gxF, nullptr, Md, G4H, IN0P, 0);
  gemm_wmma_kernel<<<dim3(Md / 16, G4H / 128), 256, 0, stream>>>(
      x0, wih0h + (size_t)G4H * IN0P, b_ih0 + G4H, b_hh0 + G4H, gxB, nullptr, Md, G4H, IN0P, 0);
  lstm_scan_kernel<<<2, 512, scanLds, stream>>>(
      gxF, gxB, whh0h, whh0h + (size_t)G4H * Hd, xb0);

  // -------- layers 1..3 (K = 512) --------
  _Float16* xin = xb0;
  _Float16* xout = xb1;
  for (int l = 0; l < 3; ++l) {
    const _Float16* wf  = wihh + (size_t)(l * 2 + 0) * G4H * H2d;
    const _Float16* wb  = wihh + (size_t)(l * 2 + 1) * G4H * H2d;
    const _Float16* rwf = whhh + (size_t)(l * 2 + 0) * G4H * Hd;
    const _Float16* rwb = whhh + (size_t)(l * 2 + 1) * G4H * Hd;
    gemm_wmma_kernel<<<dim3(Md / 16, G4H / 128), 256, 0, stream>>>(
        xin, wf, b_ih + (size_t)(l * 2 + 0) * G4H, b_hh + (size_t)(l * 2 + 0) * G4H,
        gxF, nullptr, Md, G4H, H2d, 0);
    gemm_wmma_kernel<<<dim3(Md / 16, G4H / 128), 256, 0, stream>>>(
        xin, wb, b_ih + (size_t)(l * 2 + 1) * G4H, b_hh + (size_t)(l * 2 + 1) * G4H,
        gxB, nullptr, Md, G4H, H2d, 0);
    lstm_scan_kernel<<<2, 512, scanLds, stream>>>(gxF, gxB, rwf, rwb, xout);
    _Float16* tmp = xin; xin = xout; xout = tmp;
  }
  // xin == final lstm_out (f16, [1600][512])

  // -------- projections for biaffine --------
  gemm_wmma_kernel<<<dim3(Md / 16, H2d / 128), 256, 0, stream>>>(
      xin, W1a, nullptr, nullptr, nullptr, hfirst, Md, H2d, H2d, 1);
  zero_f16_kernel<<<blocks((long)Bd * 101 * H2d), 256, 0, stream>>>(hroot, (long)Bd * 101 * H2d);
  gemm_wmma_kernel<<<dim3(Md / 16, H2d / 128), 256, 0, stream>>>(
      xin, W1b, nullptr, nullptr, nullptr, hroot, Md, H2d, H2d, 2);

  // -------- fused scorer --------
  biaffine_score_kernel<<<dim3(Ld, Bd), 224, 0, stream>>>(
      hfirst, hroot, b1, b2, W3h, b3, sen_lens, W2h, out);
}
